// GraphSAGE_39556648796479
// MI455X (gfx1250) — compile-verified, matrix operand order checked by reading, not run on hardware
//
#include <hip/hip_runtime.h>

// ---------------- problem constants (from reference) ----------------
#define N_NODES 50000
#define N_EDGES 800000
#define F_IN    128
#define HIDDEN  256
#define MID     48
#define MID_PAD 64      // pad K=48 -> 64 so WMMA K-steps of 32 work
#define EMB     128

#define WPB     8       // waves per block (256 threads, wave32)
#define MTILES  (N_NODES / 16)              // 3125 exactly
#define MBLOCKS ((MTILES + WPB - 1) / WPB)  // 391
#define LDS_PAD 8       // bf16 elements of row padding in LDS (bank spread)

typedef __bf16 bf16;
typedef __attribute__((ext_vector_type(16))) __bf16 v16bf;
typedef __attribute__((ext_vector_type(8)))  __bf16 v8bf;
typedef __attribute__((ext_vector_type(8)))  float  v8f;

// ---------------- utility kernels ----------------
__global__ void k_zero_f32(float* p, long n) {
    long i = (long)blockIdx.x * blockDim.x + threadIdx.x;
    if (i < n) p[i] = 0.0f;
}

__global__ void k_zero_bf16(bf16* p, long n) {
    long i = (long)blockIdx.x * blockDim.x + threadIdx.x;
    if (i < n) p[i] = (bf16)0.0f;
}

// fp32 [rows, Kact] row-major  ->  bf16 [rows, Kld] row-major, zero-padded
__global__ void k_f32_to_bf16_pad(const float* __restrict__ in, bf16* __restrict__ out,
                                  int rows, int Kact, int Kld) {
    long i = (long)blockIdx.x * blockDim.x + threadIdx.x;
    long total = (long)rows * Kld;
    if (i >= total) return;
    int r = (int)(i / Kld);
    int k = (int)(i % Kld);
    out[i] = (k < Kact) ? (bf16)in[(long)r * Kact + k] : (bf16)0.0f;
}

// degree[dst] += 1 per edge
__global__ void k_degree(const int* __restrict__ dst, float* __restrict__ deg, int E) {
    int e = blockIdx.x * blockDim.x + threadIdx.x;
    if (e < E) atomicAdd(&deg[dst[e]], 1.0f);
}

// agg[dst, :Kact] += h[src, :Kact]   (h, agg strided by Kld)
__global__ void k_scatter(const int* __restrict__ src, const int* __restrict__ dst,
                          const bf16* __restrict__ h, float* __restrict__ agg,
                          int Kact, int Kld, int E) {
    long tid = (long)blockIdx.x * blockDim.x + threadIdx.x;
    int cpe = Kact >> 2;                       // 4 features per thread
    long total = (long)E * cpe;
    if (tid >= total) return;
    int e = (int)(tid / cpe);
    int c = (int)(tid % cpe) << 2;
    int s = src[e], d = dst[e];
    const bf16* hp = h + (long)s * Kld + c;
    float* ap = agg + (long)d * Kld + c;
    bf16 v0 = hp[0], v1 = hp[1], v2 = hp[2], v3 = hp[3];
    atomicAdd(ap + 0, (float)v0);
    atomicAdd(ap + 1, (float)v1);
    atomicAdd(ap + 2, (float)v2);
    atomicAdd(ap + 3, (float)v3);
}

// mean_bf16 = agg / max(deg, 1)   (pad columns are already zero in agg)
__global__ void k_normalize(const float* __restrict__ agg, const float* __restrict__ deg,
                            bf16* __restrict__ mean, int Kld, long total) {
    long i = (long)blockIdx.x * blockDim.x + threadIdx.x;
    if (i >= total) return;
    int n = (int)(i / Kld);
    float d = deg[n];
    d = d > 1.0f ? d : 1.0f;
    mean[i] = (bf16)(agg[i] / d);
}

// ---------------- WMMA fused SAGE GEMM (weights staged in LDS) ----------------
// out[n, o] = relu?( sum_k mean[n,k]*Wl[o,k] + sum_k h[n,k]*Wr[o,k] + bias[o] )
// One block: 8 consecutive M-tiles x one (16*NT)-wide N strip.
// The Wl/Wr strips (identical for all 8 waves) are copied to LDS once, then
// WMMA B fragments come from ds_load while A streams from global, so B waits
// ride DScnt and A loads can pipeline on LOADcnt.
// A-fragment layout (ISA 7.12.2, 16-bit A 16x32): lane L<16 holds row M=L,
// K = {k0..k0+7, k0+16..k0+23}; lane L>=16 holds row M=L-16, K = {+8, +24}.
// B mirrors A with lane index = output column.
__device__ __forceinline__ v16bf load_frag(const bf16* base, int k) {
    v8bf lo = *(const v8bf*)(base + k);
    v8bf hi = *(const v8bf*)(base + k + 16);
    v16bf v;
#pragma unroll
    for (int i = 0; i < 8; ++i) { v[i] = lo[i]; v[i + 8] = hi[i]; }
    return v;
}

template <int NT>
__global__ void k_sage_gemm(const bf16* __restrict__ A1,   // mean  [N_NODES, lda]
                            const bf16* __restrict__ A2,   // h     [N_NODES, lda]
                            const bf16* __restrict__ B1,   // Wl    [Fout, K] bf16 (lda==K)
                            const bf16* __restrict__ B2,   // Wr    [Fout, K] bf16
                            const float* __restrict__ bias,
                            bf16* __restrict__ outBf, int ldoBf,
                            float* __restrict__ outF, int ldoF,
                            int lda, int K, int applyRelu) {
    extern __shared__ bf16 smem[];
    const int wave = threadIdx.x >> 5;
    const int lane = threadIdx.x & 31;
    const int half = lane >> 4;         // 0: lanes 0-15, 1: lanes 16-31
    const int idx  = lane & 15;

    const int mBlock = blockIdx.x % MBLOCKS;
    const int tg     = blockIdx.x / MBLOCKS;
    const int n0     = tg * 16 * NT;            // first output column of strip
    const int rows   = NT * 16;                 // weight rows in strip
    const int ldsK   = K + LDS_PAD;             // padded LDS row stride

    bf16* sWl = smem;
    bf16* sWr = smem + (long)rows * ldsK;

    // cooperative stage of both weight strips into LDS (global rows contiguous)
    {
        const bf16* gWl = B1 + (long)n0 * K;
        const bf16* gWr = B2 + (long)n0 * K;
        const long total = (long)rows * K;
        for (long i = (long)threadIdx.x * 8; i < total; i += (long)blockDim.x * 8) {
            int r = (int)(i / K);
            int c = (int)(i % K);
            *(v8bf*)(sWl + (long)r * ldsK + c) = *(const v8bf*)(gWl + i);
            *(v8bf*)(sWr + (long)r * ldsK + c) = *(const v8bf*)(gWr + i);
        }
    }
    __syncthreads();

    const int tm = mBlock * WPB + wave;
    if (tm >= MTILES) return;                   // wave-uniform exit (after barrier)

    const bf16* a1 = A1 + (long)(tm * 16 + idx) * lda + half * 8;
    const bf16* a2 = A2 + (long)(tm * 16 + idx) * lda + half * 8;
    const bf16* sb1[NT];
    const bf16* sb2[NT];
#pragma unroll
    for (int j = 0; j < NT; ++j) {
        sb1[j] = sWl + (long)(j * 16 + idx) * ldsK + half * 8;
        sb2[j] = sWr + (long)(j * 16 + idx) * ldsK + half * 8;
    }

    v8f acc[NT];
#pragma unroll
    for (int j = 0; j < NT; ++j) acc[j] = (v8f){};

    for (int k = 0; k < K; k += 32) {
        v16bf av = load_frag(a1, k);
        v16bf bv[NT];
#pragma unroll
        for (int j = 0; j < NT; ++j) bv[j] = load_frag(sb1[j], k);
#pragma unroll
        for (int j = 0; j < NT; ++j)
            acc[j] = __builtin_amdgcn_wmma_f32_16x16x32_bf16(
                false, av, false, bv[j], (short)0, acc[j], false, false);
    }
    for (int k = 0; k < K; k += 32) {
        v16bf av = load_frag(a2, k);
        v16bf bv[NT];
#pragma unroll
        for (int j = 0; j < NT; ++j) bv[j] = load_frag(sb2[j], k);
#pragma unroll
        for (int j = 0; j < NT; ++j)
            acc[j] = __builtin_amdgcn_wmma_f32_16x16x32_bf16(
                false, av, false, bv[j], (short)0, acc[j], false, false);
    }

    // C/D layout: VGPR r -> row = tm*16 + r + half*8, col = n0 + j*16 + idx
#pragma unroll
    for (int j = 0; j < NT; ++j) {
        const int col = n0 + j * 16 + idx;
        const float bval = bias[col];
#pragma unroll
        for (int r = 0; r < 8; ++r) {
            int row = tm * 16 + r + half * 8;
            float v = acc[j][r] + bval;
            if (applyRelu) v = v > 0.0f ? v : 0.0f;
            if (outF)  outF[(long)row * ldoF + col] = v;
            if (outBf) outBf[(long)row * ldoBf + col] = (bf16)v;
        }
    }
}

// ---------------- host side ----------------
static inline unsigned cdiv(long a, long b) { return (unsigned)((a + b - 1) / b); }
static inline size_t align256(size_t x) { return (x + 255) & ~(size_t)255; }
static inline size_t lds_bytes(int NT, int K) {
    return (size_t)(NT * 16) * (K + LDS_PAD) * 2 /*bf16*/ * 2 /*Wl+Wr*/;
}

extern "C" void kernel_launch(void* const* d_in, const int* in_sizes, int n_in,
                              void* d_out, int out_size, void* d_ws, size_t ws_size,
                              hipStream_t stream) {
    (void)in_sizes; (void)n_in; (void)out_size; (void)ws_size;

    const float* x   = (const float*)d_in[0];
    const int*   ei  = (const int*)d_in[1];
    const int*   src = ei;
    const int*   dst = ei + N_EDGES;
    const float* W1l = (const float*)d_in[2];
    const float* W1r = (const float*)d_in[3];
    const float* b1  = (const float*)d_in[4];
    const float* W2l = (const float*)d_in[5];
    const float* W2r = (const float*)d_in[6];
    const float* b2  = (const float*)d_in[7];
    const float* W3l = (const float*)d_in[8];
    const float* W3r = (const float*)d_in[9];
    const float* b3  = (const float*)d_in[10];
    float* out = (float*)d_out;

    // workspace carve-up
    char* ws = (char*)d_ws;
    float* deg  = (float*)ws;                 ws += align256((size_t)N_NODES * 4);
    float* agg  = (float*)ws;                 ws += align256((size_t)N_NODES * HIDDEN * 4);
    bf16*  xbf  = (bf16*)ws;                  ws += align256((size_t)N_NODES * F_IN * 2);
    bf16*  h1   = (bf16*)ws;                  ws += align256((size_t)N_NODES * HIDDEN * 2);
    bf16*  h2   = (bf16*)ws;                  ws += align256((size_t)N_NODES * MID_PAD * 2);
    bf16*  mn   = (bf16*)ws;                  ws += align256((size_t)N_NODES * HIDDEN * 2);
    bf16*  wl   = (bf16*)ws;                  ws += align256((size_t)HIDDEN * HIDDEN * 2);
    bf16*  wr   = (bf16*)ws;                  ws += align256((size_t)HIDDEN * HIDDEN * 2);

    const int B = 256;

    // degree (layer-independent)
    k_zero_f32<<<cdiv(N_NODES, B), B, 0, stream>>>(deg, N_NODES);
    k_degree<<<cdiv(N_EDGES, B), B, 0, stream>>>(dst, deg, N_EDGES);

    // x -> bf16
    k_f32_to_bf16_pad<<<cdiv((long)N_NODES * F_IN, B), B, 0, stream>>>(x, xbf, N_NODES, F_IN, F_IN);

    // ---------------- layer 1: K=128 -> F=256, relu, NT=4 ----------------
    k_zero_f32<<<cdiv((long)N_NODES * F_IN, B), B, 0, stream>>>(agg, (long)N_NODES * F_IN);
    k_scatter<<<cdiv((long)N_EDGES * (F_IN / 4), B), B, 0, stream>>>(src, dst, xbf, agg, F_IN, F_IN, N_EDGES);
    k_normalize<<<cdiv((long)N_NODES * F_IN, B), B, 0, stream>>>(agg, deg, mn, F_IN, (long)N_NODES * F_IN);
    k_f32_to_bf16_pad<<<cdiv((long)HIDDEN * F_IN, B), B, 0, stream>>>(W1l, wl, HIDDEN, F_IN, F_IN);
    k_f32_to_bf16_pad<<<cdiv((long)HIDDEN * F_IN, B), B, 0, stream>>>(W1r, wr, HIDDEN, F_IN, F_IN);
    k_sage_gemm<4><<<MBLOCKS * (HIDDEN / 64), 256, lds_bytes(4, F_IN), stream>>>(
        mn, xbf, wl, wr, b1, h1, HIDDEN, nullptr, 0, F_IN, F_IN, 1);

    // ---------------- layer 2: K=256 -> F=48 (pad 64), relu, NT=3 ----------------
    k_zero_bf16<<<cdiv((long)N_NODES * MID_PAD, B), B, 0, stream>>>(h2, (long)N_NODES * MID_PAD);
    k_zero_f32<<<cdiv((long)N_NODES * HIDDEN, B), B, 0, stream>>>(agg, (long)N_NODES * HIDDEN);
    k_scatter<<<cdiv((long)N_EDGES * (HIDDEN / 4), B), B, 0, stream>>>(src, dst, h1, agg, HIDDEN, HIDDEN, N_EDGES);
    k_normalize<<<cdiv((long)N_NODES * HIDDEN, B), B, 0, stream>>>(agg, deg, mn, HIDDEN, (long)N_NODES * HIDDEN);
    k_f32_to_bf16_pad<<<cdiv((long)MID * HIDDEN, B), B, 0, stream>>>(W2l, wl, MID, HIDDEN, HIDDEN);
    k_f32_to_bf16_pad<<<cdiv((long)MID * HIDDEN, B), B, 0, stream>>>(W2r, wr, MID, HIDDEN, HIDDEN);
    k_sage_gemm<3><<<MBLOCKS * 1, 256, lds_bytes(3, HIDDEN), stream>>>(
        mn, h1, wl, wr, b2, h2, MID_PAD, nullptr, 0, HIDDEN, HIDDEN, 1);

    // ---------------- layer 3: K=48 (pad 64) -> F=128, no relu, fp32 out, NT=4 ----------------
    k_zero_f32<<<cdiv((long)N_NODES * MID_PAD, B), B, 0, stream>>>(agg, (long)N_NODES * MID_PAD);
    k_scatter<<<cdiv((long)N_EDGES * (MID / 4), B), B, 0, stream>>>(src, dst, h2, agg, MID, MID_PAD, N_EDGES);
    k_normalize<<<cdiv((long)N_NODES * MID_PAD, B), B, 0, stream>>>(agg, deg, mn, MID_PAD, (long)N_NODES * MID_PAD);
    k_f32_to_bf16_pad<<<cdiv((long)EMB * MID_PAD, B), B, 0, stream>>>(W3l, wl, EMB, MID, MID_PAD);
    k_f32_to_bf16_pad<<<cdiv((long)EMB * MID_PAD, B), B, 0, stream>>>(W3r, wr, EMB, MID, MID_PAD);
    k_sage_gemm<4><<<MBLOCKS * (EMB / 64), 256, lds_bytes(4, MID_PAD), stream>>>(
        mn, h2, wl, wr, b3, nullptr, 0, out, EMB, MID_PAD, MID_PAD, 0);
}
